// PanguModel_Plasim_45586782879915
// MI455X (gfx1250) — compile-verified, matrix operand order checked by reading, not run on hardware
//
#include <hip/hip_runtime.h>
#include <hip/hip_bf16.h>

typedef __attribute__((ext_vector_type(16))) _Float16 v16h;
typedef __attribute__((ext_vector_type(8)))  _Float16 v8h;
typedef __attribute__((ext_vector_type(8)))  float    v8f;

#define DIMC   192
#define NHEAD  6
#define HD     32
#define NWIN   1024      // nLon(16) * nW(64)
#define NTOK   144       // tokens per window
#define LTOT   147456
#define HIDDEN 768

// ---------------------------------------------------------------------------
// WMMA fragment helpers (gfx1250 wave32, 16x16x32 f16 -> f32)
// 16-bit A 16x32 lane layout: lane l holds row (l&15); half e maps to
// K = 16*(e>>3) + 8*(l>>4) + (e&7)  => two contiguous 8-half (16B) chunks.
// B loaded from N-major storage (i.e. W^T [N][K]) uses the identical pattern.
// ---------------------------------------------------------------------------
union AB16 { v16h v; v8h h[2]; };

__device__ __forceinline__ v16h load_frag(const _Float16* base, int ld, int lane) {
  int m = lane & 15, g = lane >> 4;
  const _Float16* p = base + m * ld + 8 * g;
  AB16 u;
  u.h[0] = *(const v8h*)(p);
  u.h[1] = *(const v8h*)(p + 16);
  return u.v;
}

__device__ __forceinline__ v8f wmma32(v16h a, v16h b, v8f c) {
  return __builtin_amdgcn_wmma_f32_16x16x32_f16(false, a, false, b, (short)0, c, false, false);
}

__device__ __forceinline__ v8f zero8() {
  v8f z = {0.f,0.f,0.f,0.f,0.f,0.f,0.f,0.f};
  return z;
}

__device__ __forceinline__ float wave_red_add(float v) {
  #pragma unroll
  for (int m = 16; m >= 1; m >>= 1) v += __shfl_xor(v, m, 32);
  return v;
}
__device__ __forceinline__ float half_red_max(float v) {
  #pragma unroll
  for (int m = 8; m >= 1; m >>= 1) v = fmaxf(v, __shfl_xor(v, m, 32));
  return v;
}
__device__ __forceinline__ float half_red_add(float v) {
  #pragma unroll
  for (int m = 8; m >= 1; m >>= 1) v += __shfl_xor(v, m, 32);
  return v;
}

// shifted-window region id (replicates _shift_window_mask region labeling)
__device__ __forceinline__ int region_of(int t, int pb, int latb, int lb) {
  int p = t / 72; int r = t % 72; int la = r / 12; int lo = r % 12;
  int s0 = (pb == 3)  ? ((p  >= 1) ? 2 : 1) : 0;
  int s1 = (latb == 15) ? ((la >= 3) ? 2 : 1) : 0;
  int s2 = (lb == 15 && lo >= 6) ? 1 : 0;
  return s0 * 9 + s1 * 3 + s2;
}

// window position -> original token index (roll by +SHIFT folded in)
__device__ __forceinline__ int src_token(int widx, int n) {
  int lb = widx >> 6, w = widx & 63, pb = w >> 4, latb = w & 15;
  int p = n / 72, r = n % 72, la = r / 12, lo = r % 12;
  int z = pb * 2 + p + 1;   if (z >= 8)   z -= 8;
  int y = latb * 6 + la + 3; if (y >= 96)  y -= 96;
  int x = lb * 12 + lo + 6;  if (x >= 192) x -= 192;
  return (z * 96 + y) * 192 + x;
}

// ---------------------------------------------------------------------------
// Weight convert+transpose: W[K][N] f32 -> Wt[N][K] f16
// ---------------------------------------------------------------------------
__global__ void k_wt(const float* __restrict__ W, _Float16* __restrict__ Wt, int K, int N) {
  int i = blockIdx.x * blockDim.x + threadIdx.x;
  if (i >= K * N) return;
  int n = i / K, k = i % K;
  Wt[n * K + k] = (_Float16)W[k * N + n];
}

// earth-position bias pre-gather: pre[h][w][i][j] = tbl[epi(i,j)][w][h]
__global__ void k_epb(const float* __restrict__ tbl, float* __restrict__ pre) {
  int i = blockIdx.x * blockDim.x + threadIdx.x;
  if (i >= NHEAD * 64 * NTOK * NTOK) return;
  int j  = i % NTOK;
  int t  = (i / NTOK) % NTOK;
  int w  = (i / (NTOK * NTOK)) % 64;
  int h  = i / (NTOK * NTOK * 64);
  int pi = t / 72, li = (t % 72) / 12, xi = t % 12;
  int pj = j / 72, lj = (j % 72) / 12, xj = j % 12;
  int epi = 828 * (pi + 2 * pj) + 23 * (li + 6 * lj) + (xi - xj + 11);
  pre[i] = tbl[epi * (64 * NHEAD) + w * NHEAD + h];
}

__global__ void k_zero(_Float16* __restrict__ p, int n) {
  int i = (blockIdx.x * blockDim.x + threadIdx.x) * 8;
  if (i + 7 >= n) return;
  v8h z;
  #pragma unroll
  for (int j = 0; j < 8; ++j) z[j] = (_Float16)0.f;
  *(v8h*)(p + i) = z;
}

// ---------------------------------------------------------------------------
// LN1 + roll + window partition -> h_norm f16 [NWIN][144][192]
// one wave per output token, 8 waves / block
// ---------------------------------------------------------------------------
__global__ void __launch_bounds__(256) k_ln1(const float* __restrict__ x,
                                             const float* __restrict__ g1,
                                             const float* __restrict__ b1,
                                             _Float16* __restrict__ hb) {
  int wave = threadIdx.x >> 5, lane = threadIdx.x & 31;
  int tout = blockIdx.x * 8 + wave;        // [0, 147456)
  int widx = tout / NTOK, n = tout % NTOK;
  long long src = (long long)src_token(widx, n) * DIMC;
  float v[6]; float s = 0.f;
  #pragma unroll
  for (int j = 0; j < 6; ++j) { v[j] = x[src + lane + 32 * j]; s += v[j]; }
  s = wave_red_add(s);
  float mu = s * (1.f / 192.f);
  float q = 0.f;
  #pragma unroll
  for (int j = 0; j < 6; ++j) { float d = v[j] - mu; q += d * d; }
  q = wave_red_add(q);
  float rstd = rsqrtf(q * (1.f / 192.f) + 1e-5f);
  #pragma unroll
  for (int j = 0; j < 6; ++j) {
    int c = lane + 32 * j;
    hb[(long long)tout * DIMC + c] = (_Float16)((v[j] - mu) * rstd * g1[c] + b1[c]);
  }
}

// ---------------------------------------------------------------------------
// QKV GEMM per window: [144,192] x [192,576]
// nt split into three uniform ranges (q / k / v) so sel & head are scalar:
//   nt in [0,12): q,  h = nt>>1, d = (nt&1)*16 + lane
//   nt in [0,12): k   (weight col offset 192)
//   nt in [0,12): v   (weight col offset 384, stored transposed + K-padded)
// ---------------------------------------------------------------------------
__global__ void __launch_bounds__(288) k_qkv(const _Float16* __restrict__ hb,
                                             const _Float16* __restrict__ wt,
                                             const float* __restrict__ bias,
                                             _Float16* __restrict__ q,
                                             _Float16* __restrict__ kbuf,
                                             _Float16* __restrict__ vt) {
  int widx = blockIdx.x;
  int s = threadIdx.x >> 5, lane = threadIdx.x & 31, g = lane >> 4;
  int ln = lane & 15;
  const _Float16* abase = hb + (long long)widx * NTOK * DIMC + s * 16 * DIMC;
  v16h A[6];
  #pragma unroll
  for (int kt = 0; kt < 6; ++kt) A[kt] = load_frag(abase + kt * 32, DIMC, lane);

  // ---- Q (scaled) ----
  for (int nt = 0; nt < 12; ++nt) {
    v8f acc = zero8();
    #pragma unroll
    for (int kt = 0; kt < 6; ++kt) {
      v16h B = load_frag(wt + (nt * 16) * DIMC + kt * 32, DIMC, lane);
      acc = wmma32(A[kt], B, acc);
    }
    int h = nt >> 1;
    int d = ((nt & 1) << 4) + ln;
    float bv = bias[nt * 16 + ln];
    long long base = ((long long)(widx * NHEAD + h) * NTOK + s * 16 + 8 * g) * HD + d;
    #pragma unroll
    for (int r = 0; r < 8; ++r)
      q[base + (long long)r * HD] = (_Float16)((acc[r] + bv) * 0.17677669529663687f);
  }
  // ---- K ----
  for (int nt = 0; nt < 12; ++nt) {
    v8f acc = zero8();
    #pragma unroll
    for (int kt = 0; kt < 6; ++kt) {
      v16h B = load_frag(wt + ((nt + 12) * 16) * DIMC + kt * 32, DIMC, lane);
      acc = wmma32(A[kt], B, acc);
    }
    int h = nt >> 1;
    int d = ((nt & 1) << 4) + ln;
    float bv = bias[192 + nt * 16 + ln];
    long long base = ((long long)(widx * NHEAD + h) * NTOK + s * 16 + 8 * g) * HD + d;
    #pragma unroll
    for (int r = 0; r < 8; ++r)
      kbuf[base + (long long)r * HD] = (_Float16)(acc[r] + bv);
  }
  // ---- V (transposed: vt[w][h][d][i], K-padded to 160) ----
  for (int nt = 0; nt < 12; ++nt) {
    v8f acc = zero8();
    #pragma unroll
    for (int kt = 0; kt < 6; ++kt) {
      v16h B = load_frag(wt + ((nt + 24) * 16) * DIMC + kt * 32, DIMC, lane);
      acc = wmma32(A[kt], B, acc);
    }
    int h = nt >> 1;
    int d = ((nt & 1) << 4) + ln;
    float bv = bias[384 + nt * 16 + ln];
    long long base = ((long long)(widx * NHEAD + h) * HD + d) * 160 + s * 16 + 8 * g;
    #pragma unroll
    for (int r = 0; r < 8; ++r)
      vt[base + r] = (_Float16)(acc[r] + bv);
  }
}

// ---------------------------------------------------------------------------
// Fused attention per (window, head): S = qk^T + bias; softmax; O = P v
// ---------------------------------------------------------------------------
__global__ void __launch_bounds__(288) k_attn(const _Float16* __restrict__ q,
                                              const _Float16* __restrict__ kk,
                                              const _Float16* __restrict__ vt,
                                              const float* __restrict__ epb,
                                              _Float16* __restrict__ ob) {
  __shared__ _Float16 P[9][16][160];
  int widx = blockIdx.x / NHEAD, h = blockIdx.x % NHEAD;
  int lb = widx >> 6, w = widx & 63, pb = w >> 4, latb = w & 15;
  int s = threadIdx.x >> 5, lane = threadIdx.x & 31, g = lane >> 4;
  int i0 = s * 16;
  long long hb = (long long)(widx * NHEAD + h);

  v16h Aq = load_frag(q + hb * NTOK * HD + i0 * HD, HD, lane);
  v8f c[9];
  #pragma unroll
  for (int t = 0; t < 9; ++t) {
    v16h B = load_frag(kk + hb * NTOK * HD + (t * 16) * HD, HD, lane);
    c[t] = wmma32(Aq, B, zero8());
  }
  // bias: dense pre-gathered earth-position bias + arithmetic shift-mask
  int rreg[8];
  #pragma unroll
  for (int r = 0; r < 8; ++r) rreg[r] = region_of(i0 + 8 * g + r, pb, latb, lb);
  const float* eb = epb + ((long long)(h * 64 + w) * NTOK) * NTOK;
  #pragma unroll
  for (int t = 0; t < 9; ++t) {
    int n = t * 16 + (lane & 15);
    int creg = region_of(n, pb, latb, lb);
    #pragma unroll
    for (int r = 0; r < 8; ++r) {
      int i = i0 + 8 * g + r;
      c[t][r] += eb[(long long)i * NTOK + n] + ((rreg[r] != creg) ? -100.f : 0.f);
    }
  }
  // softmax over the 144-wide row (row lives in 16 lanes of one half-wave)
  #pragma unroll
  for (int r = 0; r < 8; ++r) {
    float mx = -3.4e38f;
    #pragma unroll
    for (int t = 0; t < 9; ++t) mx = fmaxf(mx, c[t][r]);
    mx = half_red_max(mx);
    float sum = 0.f;
    #pragma unroll
    for (int t = 0; t < 9; ++t) { float e = __expf(c[t][r] - mx); c[t][r] = e; sum += e; }
    sum = half_red_add(sum);
    float inv = 1.f / sum;
    #pragma unroll
    for (int t = 0; t < 9; ++t)
      P[s][8 * g + r][t * 16 + (lane & 15)] = (_Float16)(c[t][r] * inv);
    P[s][8 * g + r][144 + (lane & 15)] = (_Float16)0.f;   // K-pad
  }
  __syncthreads();
  // O(16x32) = P(16x160) @ Vt(160x32), K padded to 5x32
  #pragma unroll
  for (int ntile = 0; ntile < 2; ++ntile) {
    v8f oc = zero8();
    #pragma unroll
    for (int kt = 0; kt < 5; ++kt) {
      v16h Ap = load_frag(&P[s][0][0] + kt * 32, 160, lane);
      v16h B  = load_frag(vt + (hb * HD + ntile * 16) * 160 + kt * 32, 160, lane);
      oc = wmma32(Ap, B, oc);
    }
    int d = ntile * 16 + (lane & 15);
    #pragma unroll
    for (int r = 0; r < 8; ++r) {
      int i = i0 + 8 * g + r;
      ob[((long long)widx * NTOK + i) * DIMC + h * HD + d] = (_Float16)oc[r];
    }
  }
}

// ---------------------------------------------------------------------------
// proj GEMM + window-reverse + roll-back + residual: y = x + o @ Wp + bp
// ---------------------------------------------------------------------------
__global__ void __launch_bounds__(288) k_proj(const _Float16* __restrict__ ob,
                                              const _Float16* __restrict__ wt,
                                              const float* __restrict__ bias,
                                              const float* __restrict__ x,
                                              float* __restrict__ y) {
  int widx = blockIdx.x;
  int s = threadIdx.x >> 5, lane = threadIdx.x & 31, g = lane >> 4;
  int i0 = s * 16;
  const _Float16* abase = ob + ((long long)widx * NTOK + i0) * DIMC;
  v16h A[6];
  #pragma unroll
  for (int kt = 0; kt < 6; ++kt) A[kt] = load_frag(abase + kt * 32, DIMC, lane);
  int srcT[8];
  #pragma unroll
  for (int r = 0; r < 8; ++r) srcT[r] = src_token(widx, i0 + 8 * g + r);
  for (int nt = 0; nt < 12; ++nt) {
    v8f acc = zero8();
    #pragma unroll
    for (int kt = 0; kt < 6; ++kt) {
      v16h B = load_frag(wt + (nt * 16) * DIMC + kt * 32, DIMC, lane);
      acc = wmma32(A[kt], B, acc);
    }
    int cc = nt * 16 + (lane & 15);
    float bv = bias[cc];
    #pragma unroll
    for (int r = 0; r < 8; ++r) {
      long long idx = (long long)srcT[r] * DIMC + cc;
      y[idx] = x[idx] + acc[r] + bv;
    }
  }
}

// ---------------------------------------------------------------------------
// Fused LN2 + FC1 + GELU + FC2 + residual; hidden strip kept in LDS
// 4 waves/block, 16 tokens/wave, 30KB LDS per wave (dynamic).
// Phase B runs kt-outer with 12 live accumulators: each LDS A tile loaded once.
// ---------------------------------------------------------------------------
__global__ void __launch_bounds__(128) k_mlp(const float* __restrict__ y,
                                             const float* __restrict__ g2,
                                             const float* __restrict__ b2,
                                             const _Float16* __restrict__ w1t,
                                             const float* __restrict__ b1,
                                             const _Float16* __restrict__ w2t,
                                             const float* __restrict__ bb2,
                                             float* __restrict__ out) {
  extern __shared__ _Float16 smem[];
  int wave = threadIdx.x >> 5, lane = threadIdx.x & 31, g = lane >> 4;
  _Float16* h2  = smem + wave * 15360;       // [16][192]
  _Float16* hid = h2 + 16 * DIMC;            // [16][768]
  long long tok0 = (long long)blockIdx.x * 64 + wave * 16;

  // LN2 (one full-wave reduction per token)
  for (int rr = 0; rr < 16; ++rr) {
    long long t = tok0 + rr;
    float v[6]; float ssum = 0.f;
    #pragma unroll
    for (int j = 0; j < 6; ++j) { v[j] = y[t * DIMC + lane + 32 * j]; ssum += v[j]; }
    ssum = wave_red_add(ssum);
    float mu = ssum * (1.f / 192.f);
    float qv = 0.f;
    #pragma unroll
    for (int j = 0; j < 6; ++j) { float d = v[j] - mu; qv += d * d; }
    qv = wave_red_add(qv);
    float rstd = rsqrtf(qv * (1.f / 192.f) + 1e-5f);
    #pragma unroll
    for (int j = 0; j < 6; ++j) {
      int cc = lane + 32 * j;
      h2[rr * DIMC + cc] = (_Float16)((v[j] - mu) * rstd * g2[cc] + b2[cc]);
    }
  }
  __syncthreads();

  // FC1 + exact GELU -> hidden in LDS (A tiles register-resident)
  v16h A[6];
  #pragma unroll
  for (int kt = 0; kt < 6; ++kt) A[kt] = load_frag(h2 + kt * 32, DIMC, lane);
  for (int nt = 0; nt < 48; ++nt) {
    v8f acc = zero8();
    #pragma unroll
    for (int kt = 0; kt < 6; ++kt) {
      v16h B = load_frag(w1t + (nt * 16) * DIMC + kt * 32, DIMC, lane);
      acc = wmma32(A[kt], B, acc);
    }
    int cc = nt * 16 + (lane & 15);
    float bv = b1[cc];
    #pragma unroll
    for (int r = 0; r < 8; ++r) {
      float u = acc[r] + bv;
      float ge = u * 0.5f * (1.f + erff(u * 0.70710678118f));
      hid[(8 * g + r) * HIDDEN + cc] = (_Float16)ge;
    }
  }
  __syncthreads();

  // FC2 + residual, kt-outer: A tile from LDS loaded once per kt
  v8f acc2[12];
  #pragma unroll
  for (int nt = 0; nt < 12; ++nt) acc2[nt] = zero8();
  for (int kt = 0; kt < 24; ++kt) {
    v16h Ah = load_frag(hid + kt * 32, HIDDEN, lane);
    #pragma unroll
    for (int nt = 0; nt < 12; ++nt) {
      v16h B = load_frag(w2t + (nt * 16) * HIDDEN + kt * 32, HIDDEN, lane);
      acc2[nt] = wmma32(Ah, B, acc2[nt]);
    }
  }
  #pragma unroll
  for (int nt = 0; nt < 12; ++nt) {
    int cc = nt * 16 + (lane & 15);
    float bv = bb2[cc];
    #pragma unroll
    for (int r = 0; r < 8; ++r) {
      long long t = tok0 + 8 * g + r;
      out[t * DIMC + cc] = y[t * DIMC + cc] + acc2[nt][r] + bv;
    }
  }
}

// ---------------------------------------------------------------------------
extern "C" void kernel_launch(void* const* d_in, const int* in_sizes, int n_in,
                              void* d_out, int out_size, void* d_ws, size_t ws_size,
                              hipStream_t stream) {
  const float* x      = (const float*)d_in[0];
  const float* n1g    = (const float*)d_in[1];
  const float* n1b    = (const float*)d_in[2];
  const float* qkv_w  = (const float*)d_in[3];
  const float* qkv_b  = (const float*)d_in[4];
  const float* epb_t  = (const float*)d_in[5];
  const float* proj_w = (const float*)d_in[6];
  const float* proj_b = (const float*)d_in[7];
  const float* n2g    = (const float*)d_in[8];
  const float* n2b    = (const float*)d_in[9];
  const float* fc1_w  = (const float*)d_in[10];
  const float* fc1_b  = (const float*)d_in[11];
  const float* fc2_w  = (const float*)d_in[12];
  const float* fc2_b  = (const float*)d_in[13];
  float* out = (float*)d_out;

  char* ws = (char*)d_ws;
  size_t off = 0;
  auto alloc = [&](size_t bytes) -> void* {
    void* p = ws + off;
    off += (bytes + 255) & ~(size_t)255;
    return p;
  };
  const size_t nwtok = (size_t)NWIN * NTOK;                  // 147456
  _Float16* hbuf    = (_Float16*)alloc(nwtok * DIMC * 2);    // h_norm, reused as o_buf
  _Float16* qb      = (_Float16*)alloc(nwtok * DIMC * 2);    // q (scaled)
  _Float16* kb      = (_Float16*)alloc(nwtok * DIMC * 2);    // k
  _Float16* vtb     = (_Float16*)alloc((size_t)NWIN * NHEAD * HD * 160 * 2); // v^T, K-padded
  float*    yb      = (float*)alloc((size_t)LTOT * DIMC * 4);
  float*    epb_pre = (float*)alloc((size_t)NHEAD * 64 * NTOK * NTOK * 4);
  _Float16* qkv_wt  = (_Float16*)alloc((size_t)576 * DIMC * 2);
  _Float16* proj_wt = (_Float16*)alloc((size_t)DIMC * DIMC * 2);
  _Float16* fc1_wt  = (_Float16*)alloc((size_t)HIDDEN * DIMC * 2);
  _Float16* fc2_wt  = (_Float16*)alloc((size_t)DIMC * HIDDEN * 2);

  // weights -> f16, N-major (W^T) for B-fragment streaming
  k_wt<<<(192 * 576 + 255) / 256, 256, 0, stream>>>(qkv_w, qkv_wt, 192, 576);
  k_wt<<<(192 * 192 + 255) / 256, 256, 0, stream>>>(proj_w, proj_wt, 192, 192);
  k_wt<<<(192 * 768 + 255) / 256, 256, 0, stream>>>(fc1_w, fc1_wt, 192, 768);
  k_wt<<<(768 * 192 + 255) / 256, 256, 0, stream>>>(fc2_w, fc2_wt, 768, 192);
  // dense bias pre-gather (31.8MB, L2-resident across the 16 lon-blocks)
  k_epb<<<(NHEAD * 64 * NTOK * NTOK + 255) / 256, 256, 0, stream>>>(epb_t, epb_pre);
  // zero v^T (covers the K-pad columns 144..159)
  {
    int nv = NWIN * NHEAD * HD * 160;
    k_zero<<<(nv / 8 + 255) / 256, 256, 0, stream>>>(vtb, nv);
  }
  // LN1 + shift-roll + window partition
  k_ln1<<<LTOT / 8, 256, 0, stream>>>(x, n1g, n1b, hbuf);
  // QKV
  k_qkv<<<NWIN, 288, 0, stream>>>(hbuf, qkv_wt, qkv_b, qb, kb, vtb);
  // attention (o_buf aliases hbuf; h_norm fully consumed by k_qkv)
  k_attn<<<NWIN * NHEAD, 288, 0, stream>>>(qb, kb, vtb, epb_pre, hbuf);
  // proj + reverse roll + residual
  k_proj<<<NWIN, 288, 0, stream>>>(hbuf, proj_wt, proj_b, x, yb);
  // fused LN2 + MLP + residual (4 waves * 30KB dynamic LDS)
  k_mlp<<<LTOT / 64, 128, 4 * 15360 * sizeof(_Float16), stream>>>(
      yb, n2g, n2b, fc1_wt, fc1_b, fc2_wt, fc2_b, out);
}
